// Grid2MeshGNN_69621419868950
// MI455X (gfx1250) — compile-verified, matrix operand order checked by reading, not run on hardware
//
#include <hip/hip_runtime.h>
#include <hip/hip_bf16.h>

#define N_GRID   262144
#define N_MESH   40962
#define N_EDGES  524288

typedef __bf16 bf16_t;
typedef bf16_t v16bf  __attribute__((ext_vector_type(16)));
typedef bf16_t bf16x2 __attribute__((ext_vector_type(2)));
typedef float  v8f    __attribute__((ext_vector_type(8)));

// gfx1250 async Global->LDS copy builtins (ASYNCcnt path), guarded so the
// file still compiles on toolchains that lack them.
#if defined(__has_builtin)
#if __has_builtin(__builtin_amdgcn_global_load_async_to_lds_b128) && \
    __has_builtin(__builtin_amdgcn_s_wait_asynccnt)
#define HAVE_GFX1250_ASYNC 1
#endif
#endif
#ifndef HAVE_GFX1250_ASYNC
#define HAVE_GFX1250_ASYNC 0
#endif

// The builtin's parameters are pointers to a 16-byte int vector (GCC vector
// style), in AS1 (global) and AS3 (LDS) respectively.
typedef int v4i __attribute__((vector_size(16)));
typedef __attribute__((address_space(1))) v4i as1_v4i;
typedef __attribute__((address_space(3))) v4i as3_v4i;

// 16-byte Global->LDS copy. Async variant bypasses VGPRs entirely and is
// tracked on ASYNCcnt; on CDNA5 the low 32 bits of a generic LDS pointer are
// the LDS byte address (ISA 10.2 aperture mapping), and global addresses are
// identical in AS1 vs flat, so the integer-casts below are value-preserving.
__device__ __forceinline__ void copy16_g2l(const void* g, void* l) {
#if HAVE_GFX1250_ASYNC
  __builtin_amdgcn_global_load_async_to_lds_b128(
      (as1_v4i*)(unsigned long long)g,
      (as3_v4i*)(unsigned int)(unsigned long long)l, 0, 0);
#else
  *(int4*)l = *(const int4*)g;
#endif
}

__device__ __forceinline__ void wait_async_copies() {
#if HAVE_GFX1250_ASYNC
  __builtin_amdgcn_s_wait_asynccnt(0);
#endif
}

// ---------------------------------------------------------------------------
// One 16x16xK (K multiple of 32) bf16 WMMA accumulation over a 64-wide K chunk.
// lA  : 16 rows of A in LDS, row-major bf16, leading dim lda (points at k-chunk)
// lWt : 16 cols of B stored transposed in LDS: lWt[n*ldw + k], bf16
// A-fragment lane/K layout per CDNA5 ISA 7.12.2 (16-bit A 16x32):
//   lane group g = lane>>4, VGPR i holds K pair kp(i) = (i<4 ? 2i : 2i+8) + 8g
// ---------------------------------------------------------------------------
__device__ __forceinline__ v8f wmma_chunk(const bf16_t* lA, int lda,
                                          const bf16_t* lWt, int ldw, v8f acc) {
  const int lane = threadIdx.x & 31;
  const int g = lane >> 4;
  const int m = lane & 15;   // A: row index, B: col index
#pragma unroll
  for (int kc = 0; kc < 64; kc += 32) {
    v16bf a, b;
#pragma unroll
    for (int i = 0; i < 8; ++i) {
      const int kp = ((i < 4) ? 2 * i : 2 * i + 8) + g * 8 + kc;
      bf16x2 ap = *(const bf16x2*)(lA + m * lda + kp);
      bf16x2 bp = *(const bf16x2*)(lWt + m * ldw + kp);
      a[2 * i] = ap.x; a[2 * i + 1] = ap.y;
      b[2 * i] = bp.x; b[2 * i + 1] = bp.y;
    }
    acc = __builtin_amdgcn_wmma_f32_16x16x32_bf16(false, a, false, b,
                                                  (short)0, acc, false, false);
  }
  return acc;
}

// ---------------------------------------------------------------------------
// Block-level GEMM layer: Y[64 x N] = A[64 x K] @ W[K x N] + b
// 4 waves, wave w owns rows 16w..16w+15. Weights streamed from global in
// transposed 64-deep K chunks into lWt (N*64 bf16), with a prefetch of the
// next chunk (global_prefetch_b8). Accumulators returned in C-fragment
// layout (N = lane&15, M = r + 8*(lane>>4)).
// ---------------------------------------------------------------------------
template <int NT>
__device__ __forceinline__ void gemm_layer(const bf16_t* lA, int lda, int K,
                                           const float* __restrict__ Wg, int N,
                                           const float* __restrict__ bg,
                                           bf16_t* lWt, float* lb, v8f (&acc)[NT]) {
  for (int i = threadIdx.x; i < N; i += blockDim.x) lb[i] = bg[i];
  __syncthreads();
  const int lane = threadIdx.x & 31;
  const int n = lane & 15;
  const int wave = threadIdx.x >> 5;
#pragma unroll
  for (int t = 0; t < NT; ++t) {
    const float bv = lb[t * 16 + n];
    v8f z;
#pragma unroll
    for (int r = 0; r < 8; ++r) z[r] = bv;
    acc[t] = z;
  }
  for (int k0 = 0; k0 < K; k0 += 64) {
    __syncthreads();
    if (k0 + 64 < K)  // warm WGP$ with next weight chunk while this one stages
      __builtin_prefetch(Wg + (size_t)(k0 + 64) * N + threadIdx.x * 16, 0, 1);
    for (int idx = threadIdx.x; idx < N * 64; idx += blockDim.x) {
      const int nn = idx >> 6, kk = idx & 63;
      lWt[nn * 64 + kk] = (bf16_t)Wg[(size_t)(k0 + kk) * N + nn];
    }
    __syncthreads();
    const bf16_t* Aw = lA + (wave * 16) * lda + k0;
#pragma unroll
    for (int t = 0; t < NT; ++t)
      acc[t] = wmma_chunk(Aw, lda, lWt + t * 16 * 64, 64, acc[t]);
  }
}

// ---------------------------------------------------------------------------
__global__ void k_zero(float* __restrict__ p, int n) {
  const int i = blockIdx.x * blockDim.x + threadIdx.x;
  if (i < n) p[i] = 0.0f;
}

// grid embed: [N_GRID,64] -> relu(xW1+b1) -> W2+b2 -> bf16 [N_GRID,128]
__global__ __launch_bounds__(128) void k_grid_embed(
    const float* __restrict__ gridf,
    const float* __restrict__ W1, const float* __restrict__ b1,
    const float* __restrict__ W2, const float* __restrict__ b2,
    bf16_t* __restrict__ gemb) {
  extern __shared__ char smem[];
  bf16_t* lX  = (bf16_t*)smem;                              // 64*64  = 8 KB
  bf16_t* lH  = (bf16_t*)(smem + 8192);                     // 64*256 = 32 KB
  bf16_t* lWt = (bf16_t*)(smem + 8192 + 32768);             // 256*64 = 32 KB
  float*  lb  = (float*) (smem + 8192 + 32768 + 32768);     // 1 KB
  const int row0 = blockIdx.x * 64;
  for (int idx = threadIdx.x; idx < 64 * 64; idx += blockDim.x) {
    const int r = idx >> 6, c = idx & 63;
    lX[r * 64 + c] = (bf16_t)gridf[(size_t)(row0 + r) * 64 + c];
  }
  v8f acc1[16];
  gemm_layer<16>(lX, 64, 64, W1, 256, b1, lWt, lb, acc1);
  const int lane = threadIdx.x & 31, g = lane >> 4, n = lane & 15, wave = threadIdx.x >> 5;
#pragma unroll
  for (int t = 0; t < 16; ++t)
#pragma unroll
    for (int r = 0; r < 8; ++r) {
      const float v = fmaxf(acc1[t][r], 0.0f);
      lH[(wave * 16 + 8 * g + r) * 256 + t * 16 + n] = (bf16_t)v;
    }
  v8f acc2[8];
  gemm_layer<8>(lH, 256, 256, W2, 128, b2, lWt, lb, acc2);
#pragma unroll
  for (int t = 0; t < 8; ++t)
#pragma unroll
    for (int r = 0; r < 8; ++r) {
      const int rr = row0 + wave * 16 + 8 * g + r;
      gemb[(size_t)rr * 128 + t * 16 + n] = (bf16_t)acc2[t][r];
    }
}

// edge MLP: gather [src(bf16 gemb) | dst(mesh) | edge] = 320 -> 256 -> 64,
// scatter-add into agg[dst-N_GRID]. The src half of the gather (pure bf16
// byte-move, the dominant HBM traffic) goes through the async Global->LDS
// path; the f32 halves are converted in VALU while those copies are in flight.
__global__ __launch_bounds__(128) void k_edge_msg(
    const long long* __restrict__ eidx,
    const float* __restrict__ meshf,
    const float* __restrict__ ef,
    const bf16_t* __restrict__ gemb,
    const float* __restrict__ W1, const float* __restrict__ b1,
    const float* __restrict__ W2, const float* __restrict__ b2,
    float* __restrict__ agg) {
  extern __shared__ char smem[];
  int*    lSid = (int*)smem;                                  // 64 * 4
  int*    lDid = (int*)(smem + 256);                          // 64 * 4
  bf16_t* lA   = (bf16_t*)(smem + 512);                       // 64*320 = 40 KB
  bf16_t* lH   = (bf16_t*)(smem + 512 + 40960);               // 32 KB
  bf16_t* lWt  = (bf16_t*)(smem + 512 + 40960 + 32768);       // 32 KB
  float*  lb   = (float*) (smem + 512 + 40960 + 32768 + 32768);
  const int e0 = blockIdx.x * 64;
  for (int i = threadIdx.x; i < 64; i += blockDim.x) {
    lSid[i] = (int)eidx[e0 + i];
    lDid[i] = (int)(eidx[N_EDGES + e0 + i] - (long long)N_GRID);
  }
  __syncthreads();
  // src node features: 64 rows x 128 bf16 = 16 x 16B chunks per row, async.
  for (int idx = threadIdx.x; idx < 64 * 16; idx += blockDim.x) {
    const int r = idx >> 4, c = idx & 15;
    copy16_g2l(gemb + (size_t)lSid[r] * 128 + c * 8, lA + r * 320 + c * 8);
  }
  // dst node + edge features need f32->bf16 conversion; overlaps the copies.
  for (int idx = threadIdx.x; idx < 64 * 192; idx += blockDim.x) {
    const int r = idx / 192, c0 = idx % 192;
    bf16_t v;
    if (c0 < 128) v = (bf16_t)meshf[(size_t)lDid[r] * 128 + c0];
    else          v = (bf16_t)ef[(size_t)(e0 + r) * 64 + (c0 - 128)];
    lA[r * 320 + 128 + c0] = v;
  }
  wait_async_copies();   // own copies done; gemm_layer's barrier publishes lA
  v8f acc1[16];
  gemm_layer<16>(lA, 320, 320, W1, 256, b1, lWt, lb, acc1);
  const int lane = threadIdx.x & 31, g = lane >> 4, n = lane & 15, wave = threadIdx.x >> 5;
#pragma unroll
  for (int t = 0; t < 16; ++t)
#pragma unroll
    for (int r = 0; r < 8; ++r) {
      const float v = fmaxf(acc1[t][r], 0.0f);
      lH[(wave * 16 + 8 * g + r) * 256 + t * 16 + n] = (bf16_t)v;
    }
  v8f acc2[4];
  gemm_layer<4>(lH, 256, 256, W2, 64, b2, lWt, lb, acc2);
#pragma unroll
  for (int t = 0; t < 4; ++t)
#pragma unroll
    for (int r = 0; r < 8; ++r) {
      const int i = wave * 16 + 8 * g + r;
      const int d = lDid[i];
      __hip_atomic_fetch_add(&agg[(size_t)d * 64 + t * 16 + n], acc2[t][r],
                             __ATOMIC_RELAXED, __HIP_MEMORY_SCOPE_AGENT);
    }
}

// mesh update: [mesh | agg] = 192 -> 256 -> 128, residual add, write output.
__global__ __launch_bounds__(128) void k_mesh_update(
    const float* __restrict__ meshf,
    const float* __restrict__ agg,
    const float* __restrict__ W1, const float* __restrict__ b1,
    const float* __restrict__ W2, const float* __restrict__ b2,
    float* __restrict__ out) {
  extern __shared__ char smem[];
  bf16_t* lA  = (bf16_t*)smem;                         // 64*192 = 24 KB
  bf16_t* lH  = (bf16_t*)(smem + 24576);               // 32 KB
  bf16_t* lWt = (bf16_t*)(smem + 24576 + 32768);       // 32 KB
  float*  lb  = (float*) (smem + 24576 + 32768 + 32768);
  const int row0 = blockIdx.x * 64;
  for (int idx = threadIdx.x; idx < 64 * 192; idx += blockDim.x) {
    const int r = idx / 192, c = idx % 192;
    const int rr = row0 + r;
    float v = 0.0f;
    if (rr < N_MESH)
      v = (c < 128) ? meshf[(size_t)rr * 128 + c] : agg[(size_t)rr * 64 + (c - 128)];
    lA[r * 192 + c] = (bf16_t)v;
  }
  v8f acc1[16];
  gemm_layer<16>(lA, 192, 192, W1, 256, b1, lWt, lb, acc1);
  const int lane = threadIdx.x & 31, g = lane >> 4, n = lane & 15, wave = threadIdx.x >> 5;
#pragma unroll
  for (int t = 0; t < 16; ++t)
#pragma unroll
    for (int r = 0; r < 8; ++r) {
      const float v = fmaxf(acc1[t][r], 0.0f);
      lH[(wave * 16 + 8 * g + r) * 256 + t * 16 + n] = (bf16_t)v;
    }
  v8f acc2[8];
  gemm_layer<8>(lH, 256, 256, W2, 128, b2, lWt, lb, acc2);
#pragma unroll
  for (int t = 0; t < 8; ++t)
#pragma unroll
    for (int r = 0; r < 8; ++r) {
      const int rr = row0 + wave * 16 + 8 * g + r;
      if (rr < N_MESH) {
        const int col = t * 16 + n;
        out[(size_t)rr * 128 + col] = meshf[(size_t)rr * 128 + col] + acc2[t][r];
      }
    }
}

// ---------------------------------------------------------------------------
extern "C" void kernel_launch(void* const* d_in, const int* in_sizes, int n_in,
                              void* d_out, int out_size, void* d_ws, size_t ws_size,
                              hipStream_t stream) {
  const float*     gridf = (const float*)d_in[0];
  const float*     meshf = (const float*)d_in[1];
  const long long* eidx  = (const long long*)d_in[2];   // int64 [2, E]
  const float*     ef    = (const float*)d_in[3];
  const float *geW1 = (const float*)d_in[4],  *geb1 = (const float*)d_in[5];
  const float *geW2 = (const float*)d_in[6],  *geb2 = (const float*)d_in[7];
  const float *emW1 = (const float*)d_in[8],  *emb1 = (const float*)d_in[9];
  const float *emW2 = (const float*)d_in[10], *emb2 = (const float*)d_in[11];
  const float *nmW1 = (const float*)d_in[12], *nmb1 = (const float*)d_in[13];
  const float *nmW2 = (const float*)d_in[14], *nmb2 = (const float*)d_in[15];

  // Workspace: bf16 grid embedding (64 MB) then f32 agg (10.5 MB).
  bf16_t* gemb = (bf16_t*)d_ws;
  float*  agg  = (float*)((char*)d_ws + (size_t)N_GRID * 128 * sizeof(bf16_t));
  float*  out  = (float*)d_out;

  const int aggN = N_MESH * 64;
  k_zero<<<(aggN + 255) / 256, 256, 0, stream>>>(agg, aggN);

  k_grid_embed<<<N_GRID / 64, 128, 74752, stream>>>(gridf, geW1, geb1, geW2, geb2, gemb);

  k_edge_msg<<<N_EDGES / 64, 128, 108032, stream>>>(eidx, meshf, ef, gemb,
                                                    emW1, emb1, emW2, emb2, agg);

  k_mesh_update<<<(N_MESH + 63) / 64, 128, 91136, stream>>>(meshf, agg,
                                                            nmW1, nmb1, nmW2, nmb2, out);
}
